// SparseUNetCore_65223373357594
// MI455X (gfx1250) — matrix operand order repaired and checked by reading, not compiled, not run on hardware
//
#include <hip/hip_runtime.h>
#include <hip/hip_bf16.h>
#include <stdint.h>

// ---------------------------------------------------------------------------
// CDNA5 (gfx1250) sparse U-Net forward, WMMA bf16 (f32 accum) implementation.
// One wave computes a 16(spatial-W) x (NC*16)(out-channel) tile; A fragments
// are loaded once per K-step and reused across all NC output chunks.
// Grid mapping is exact (no divisions, no bounds checks, EXEC all-ones).
// ---------------------------------------------------------------------------

typedef __bf16 bf16_t;
typedef bf16_t v16bf __attribute__((ext_vector_type(16)));
typedef bf16_t v8bf  __attribute__((ext_vector_type(8)));
typedef float  v8f   __attribute__((ext_vector_type(8)));

#define LEAKY_SLOPE 0.01f
#define F_LEAKY      1
#define F_RES_BEFORE 2
#define F_RES_AFTER  4

struct Step8 { short f[8]; };
struct Desc  { int n; Step8 s[27]; };

// --------------------------- device helpers --------------------------------

__device__ __forceinline__ v8bf zero8() {
  v8bf z;
#pragma unroll
  for (int i = 0; i < 8; ++i) z[i] = (bf16_t)0.0f;
  return z;
}

// Load 8 contiguous bf16 channels (16B aligned) with bounds/zero handling.
__device__ __forceinline__ v8bf load_half(const bf16_t* __restrict__ in,
                                          int Din, int Hin, int Win, int Cin,
                                          int id, int ih, int iw, int cb, int kh) {
  if (cb < 0 || (unsigned)id >= (unsigned)Din || (unsigned)ih >= (unsigned)Hin ||
      (unsigned)iw >= (unsigned)Win)
    return zero8();
  size_t base = (((size_t)id * Hin + ih) * (size_t)Win + iw) * (size_t)Cin
              + (size_t)(cb + kh * 8);
  return *(const v8bf*)(in + base);
}

// ------------------------- generic WMMA conv kernel -------------------------
// blockDim = 128 (4 waves). blockIdx.x -> W-tile (4 waves/block),
// blockIdx.y -> output row h, blockIdx.z -> output plane dp.
template <int NC>
__global__ void __launch_bounds__(128)
k_conv_wmma(const bf16_t* __restrict__ in, const bf16_t* __restrict__ wf,
            const float* __restrict__ mask, const bf16_t* __restrict__ res,
            bf16_t* __restrict__ out,
            int Din, int Hin, int Win, int Cin,
            int Hout, int Wout, int Cout,
            int sh, int sw, int osh, int osw, int ooh, int oow,
            int flags, Desc d) {
  int lane = threadIdx.x & 31;
  int wt = blockIdx.x * 4 + (threadIdx.x >> 5);
  int h  = blockIdx.y;
  int dp = blockIdx.z;

  int m  = lane & 15;                        // spatial row of A for this lane
  int kh = lane >> 4;                        // K-half select
  int wbase = (wt * 16 + m) * sw;
  int hbase = h * sh;

  v8f acc[NC];
#pragma unroll
  for (int c = 0; c < NC; ++c)
    acc[c] = (v8f){0.f, 0.f, 0.f, 0.f, 0.f, 0.f, 0.f, 0.f};

  for (int s = 0; s < d.n; ++s) {
    const short* f = d.s[s].f;               // dz0,dy0,dx0,cb0, dz1,dy1,dx1,cb1
    v8bf h0 = load_half(in, Din, Hin, Win, Cin, dp + f[0], hbase + f[1],
                        wbase + f[2], f[3], kh);
    v8bf h1 = load_half(in, Din, Hin, Win, Cin, dp + f[4], hbase + f[5],
                        wbase + f[6], f[7], kh);
    v16bf a = __builtin_shufflevector(h0, h1, 0, 1, 2, 3, 4, 5, 6, 7,
                                      8, 9, 10, 11, 12, 13, 14, 15);
    const bf16_t* wb = wf + ((size_t)s * NC) * 512 + (size_t)lane * 16;
#pragma unroll
    for (int c = 0; c < NC; ++c) {
      v16bf b = *(const v16bf*)(wb + (size_t)c * 512);
      acc[c] = __builtin_amdgcn_wmma_f32_16x16x32_bf16(
          false, a, false, b, (short)0, acc[c], false, false);
    }
  }

  // D layout: lane holds column N = lane%16, rows M = (lane>>4)*8 + r.
  int oh = h * osh + ooh;
  size_t sprow = ((size_t)dp * Hout + oh) * (size_t)Wout;
  float mv[8];
  size_t sp[8];
#pragma unroll
  for (int r = 0; r < 8; ++r) {
    int ow = (wt * 16 + kh * 8 + r) * osw + oow;
    sp[r] = sprow + ow;
    mv[r] = mask[sp[r]];
  }
#pragma unroll
  for (int c = 0; c < NC; ++c) {
    int n = c * 16 + m;
#pragma unroll
    for (int r = 0; r < 8; ++r) {
      size_t oi = sp[r] * (size_t)Cout + n;
      float y = acc[c][r] * mv[r];
      if (flags & F_RES_BEFORE) y += (float)res[oi];
      if (flags & F_LEAKY) y = (y > 0.f) ? y : LEAKY_SLOPE * y;
      if (flags & F_RES_AFTER) y += (float)res[oi];
      out[oi] = (bf16_t)y;
    }
  }
}

// -------------------- weight fragment preparation ---------------------------
// frag[(s*nc + n)*512 + lane*16 + j] = W[K][N], K=(j<8 ? (lane>>4)*8+j : 16+(lane>>4)*8+j-8), N=nc*16+lane&15
__global__ void __launch_bounds__(256)
k_prep_w(const float* __restrict__ w, bf16_t* __restrict__ frag,
         int CI, int CO, int ncCount, Desc d) {
  int idx = blockIdx.x * blockDim.x + threadIdx.x;
  int total = d.n * ncCount * 512;
  if (idx >= total) return;
  int j    = idx & 15;
  int lane = (idx >> 4) & 31;
  int t2   = idx >> 9;
  int nc   = t2 % ncCount;
  int s    = t2 / ncCount;
  int K = (j < 8) ? (lane >> 4) * 8 + j : 16 + (lane >> 4) * 8 + (j - 8);
  int N = nc * 16 + (lane & 15);
  const short* f = d.s[s].f;                 // wt0,cb0, wt1,cb1
  int wt, cb;
  if (K < 16) { wt = f[0]; cb = f[1]; }
  else        { wt = f[2]; cb = f[3]; K -= 16; }
  float v = 0.0f;
  if (cb >= 0) v = w[((size_t)wt * CI + (size_t)(cb + K)) * (size_t)CO + N];
  frag[idx] = (bf16_t)v;
}

// ----------------------------- small kernels --------------------------------
__global__ void k_pool2(const float* __restrict__ mi, float* __restrict__ mo,
                        int D, int Ho, int Wo, int Hi, int Wi) {
  int i = blockIdx.x * blockDim.x + threadIdx.x;
  int tot = D * Ho * Wo;
  if (i >= tot) return;
  int w = i % Wo, h = (i / Wo) % Ho, d = i / (Wo * Ho);
  const float* p = mi + ((size_t)d * Hi + 2 * h) * (size_t)Wi + 2 * w;
  mo[i] = fmaxf(fmaxf(p[0], p[1]), fmaxf(p[Wi], p[Wi + 1]));
}

__global__ void k_maxd(const float* __restrict__ mi, float* __restrict__ mo,
                       int D, int HW) {
  int i = blockIdx.x * blockDim.x + threadIdx.x;
  if (i >= HW) return;
  float v = mi[i];
  for (int d = 1; d < D; ++d) v = fmaxf(v, mi[(size_t)d * HW + i]);
  mo[i] = v;
}

__global__ void k_f2b(const float* __restrict__ x, bf16_t* __restrict__ y, size_t n) {
  size_t i = (size_t)blockIdx.x * blockDim.x + threadIdx.x;
  if (i < n) y[i] = (bf16_t)x[i];
}

__global__ void k_b2f(const bf16_t* __restrict__ x, float* __restrict__ y, size_t n) {
  size_t i = (size_t)blockIdx.x * blockDim.x + threadIdx.x;
  if (i < n) y[i] = (float)x[i];
}

// ------------------------------- host side ----------------------------------
extern "C" void kernel_launch(void* const* d_in, const int* in_sizes, int n_in,
                              void* d_out, int out_size, void* d_ws, size_t ws_size,
                              hipStream_t stream) {
  (void)in_sizes; (void)n_in; (void)out_size; (void)ws_size;
  const float* x  = (const float*)d_in[0];
  const float* m0 = (const float*)d_in[1];
  const float* W[26];
  for (int i = 0; i < 26; ++i) W[i] = (const float*)d_in[2 + i];
  // W index map: 0..3 l0_down, 4 l0_ds, 5..8 l1_down, 9 l1_ds, 10 merger,
  //              11..14 deep, 15 splitter, 16 l1_up_conv, 17..20 l1_up,
  //              21 l0_up_conv, 22..25 l0_up

  uint8_t* ws = (uint8_t*)d_ws;
  size_t off = 0;
  auto alloc = [&](size_t bytes) -> void* {
    void* p = ws + off;
    off += (bytes + 255) & ~(size_t)255;
    return p;
  };

  const size_t nL0 = (size_t)3 * 512 * 768 * 16;   // 18,874,368
  const size_t nL1 = (size_t)3 * 256 * 384 * 32;   //  9,437,184
  const size_t nL2 = (size_t)3 * 128 * 192 * 64;   //  4,718,592
  const size_t nDP = (size_t)128 * 192 * 96;       //  2,359,296
  const size_t nM1 = (size_t)3 * 256 * 384;
  const size_t nM2 = (size_t)3 * 128 * 192;
  const size_t nMM = (size_t)128 * 192;

  bf16_t* A0 = (bf16_t*)alloc(nL0 * 2);
  bf16_t* T0 = (bf16_t*)alloc(nL0 * 2);
  bf16_t* A1 = (bf16_t*)alloc(nL1 * 2);
  bf16_t* T1 = (bf16_t*)alloc(nL1 * 2);
  bf16_t* B2 = (bf16_t*)alloc(nL2 * 2);
  bf16_t* S2 = (bf16_t*)alloc(nL2 * 2);
  bf16_t* DP = (bf16_t*)alloc(nDP * 2);
  bf16_t* TD = (bf16_t*)alloc(nDP * 2);
  float*  M1 = (float*)alloc(nM1 * 4);
  float*  M2 = (float*)alloc(nM2 * 4);
  float*  MM = (float*)alloc(nMM * 4);

  bf16_t* fr_l0dn[4]; for (int i = 0; i < 4; ++i) fr_l0dn[i] = (bf16_t*)alloc(2560 * 2);
  bf16_t* fr_ds16 = (bf16_t*)alloc(2048 * 2);
  bf16_t* fr_l1dn[4]; for (int i = 0; i < 4; ++i) fr_l1dn[i] = (bf16_t*)alloc(9216 * 2);
  bf16_t* fr_ds32 = (bf16_t*)alloc(8192 * 2);
  bf16_t* fr_mg   = (bf16_t*)alloc(18432 * 2);
  bf16_t* fr_deep[4]; for (int i = 0; i < 4; ++i) fr_deep[i] = (bf16_t*)alloc(82944 * 2);
  bf16_t* fr_sp   = (bf16_t*)alloc(18432 * 2);
  bf16_t* fr_up1  = (bf16_t*)alloc(8192 * 2);
  bf16_t* fr_l1up[4]; for (int i = 0; i < 4; ++i) fr_l1up[i] = (bf16_t*)alloc(9216 * 2);
  bf16_t* fr_up0  = (bf16_t*)alloc(2048 * 2);
  bf16_t* fr_l0up[4]; for (int i = 0; i < 4; ++i) fr_l0up[i] = (bf16_t*)alloc(2560 * 2);

  // ------------------------- descriptor tables ------------------------------
  Desc c3_16{}, p3_16{};
  c3_16.n = p3_16.n = 5;                 // 9 taps packed as 4 pairs + 1 half
  for (int s = 0; s < 5; ++s) {
    int t0 = 2 * s, t1 = 2 * s + 1;
    short* cf = c3_16.s[s].f; short* pf = p3_16.s[s].f;
    cf[0] = 0; cf[1] = (short)(t0 / 3 - 1); cf[2] = (short)(t0 % 3 - 1); cf[3] = 0;
    if (t1 < 9) { cf[4] = 0; cf[5] = (short)(t1 / 3 - 1); cf[6] = (short)(t1 % 3 - 1); cf[7] = 0; }
    else        { cf[4] = 0; cf[5] = 0; cf[6] = 0; cf[7] = -1; }
    pf[0] = (short)t0; pf[1] = 0;
    pf[2] = (short)(t1 < 9 ? t1 : 0); pf[3] = (short)(t1 < 9 ? 0 : -1);
  }

  auto build_c3 = [](Desc& cd, Desc& pd, int KC) {
    cd.n = pd.n = 9 * KC;
    for (int t = 0; t < 9; ++t)
      for (int kc = 0; kc < KC; ++kc) {
        int s = t * KC + kc;
        short dy = (short)(t / 3 - 1), dx = (short)(t % 3 - 1);
        short c0 = (short)(kc * 32), c1 = (short)(kc * 32 + 16);
        short* cf = cd.s[s].f;
        cf[0] = 0; cf[1] = dy; cf[2] = dx; cf[3] = c0;
        cf[4] = 0; cf[5] = dy; cf[6] = dx; cf[7] = c1;
        short* pf = pd.s[s].f;
        pf[0] = (short)t; pf[1] = c0; pf[2] = (short)t; pf[3] = c1;
      }
  };
  Desc c3_32{}, p3_32{}; build_c3(c3_32, p3_32, 1);
  Desc c3_96{}, p3_96{}; build_c3(c3_96, p3_96, 3);

  Desc cds16{}, pds16{};                 // 2x2 stride-2, CI=16 (tap pairs)
  cds16.n = pds16.n = 2;
  for (int s = 0; s < 2; ++s) {
    int t0 = 2 * s, t1 = 2 * s + 1;
    short* cf = cds16.s[s].f; short* pf = pds16.s[s].f;
    cf[0] = 0; cf[1] = (short)(t0 / 2); cf[2] = (short)(t0 % 2); cf[3] = 0;
    cf[4] = 0; cf[5] = (short)(t1 / 2); cf[6] = (short)(t1 % 2); cf[7] = 0;
    pf[0] = (short)t0; pf[1] = 0; pf[2] = (short)t1; pf[3] = 0;
  }

  Desc cds32{}, pds32{};                 // 2x2 stride-2, CI=32
  cds32.n = pds32.n = 4;
  for (int t = 0; t < 4; ++t) {
    short* cf = cds32.s[t].f; short* pf = pds32.s[t].f;
    cf[0] = 0; cf[1] = (short)(t / 2); cf[2] = (short)(t % 2); cf[3] = 0;
    cf[4] = 0; cf[5] = (short)(t / 2); cf[6] = (short)(t % 2); cf[7] = 16;
    pf[0] = (short)t; pf[1] = 0; pf[2] = (short)t; pf[3] = 16;
  }

  Desc cmg{}, pmg{};                     // merger 3x1x1, CI=64
  cmg.n = pmg.n = 6;
  for (int dz = 0; dz < 3; ++dz)
    for (int kc = 0; kc < 2; ++kc) {
      int s = dz * 2 + kc; short c0 = (short)(kc * 32);
      short* cf = cmg.s[s].f;
      cf[0] = (short)dz; cf[1] = 0; cf[2] = 0; cf[3] = c0;
      cf[4] = (short)dz; cf[5] = 0; cf[6] = 0; cf[7] = (short)(c0 + 16);
      short* pf = pmg.s[s].f;
      pf[0] = (short)dz; pf[1] = c0; pf[2] = (short)dz; pf[3] = (short)(c0 + 16);
    }

  Desc psp{}, csp{};                     // splitter: prep 3 planes x 3 chunks
  psp.n = 9; csp.n = 3;
  for (int p = 0; p < 3; ++p)
    for (int kc = 0; kc < 3; ++kc) {
      short* pf = psp.s[p * 3 + kc].f;
      pf[0] = (short)p; pf[1] = (short)(kc * 32);
      pf[2] = (short)p; pf[3] = (short)(kc * 32 + 16);
    }
  for (int kc = 0; kc < 3; ++kc) {
    short* cf = csp.s[kc].f;
    cf[0] = 0; cf[1] = 0; cf[2] = 0; cf[3] = (short)(kc * 32);
    cf[4] = 0; cf[5] = 0; cf[6] = 0; cf[7] = (short)(kc * 32 + 16);
  }

  auto build_up_prep = [](Desc& pd, int KC) {  // deconv 2x2: 4 taps x KC chunks
    pd.n = 4 * KC;
    for (int tap = 0; tap < 4; ++tap)
      for (int kc = 0; kc < KC; ++kc) {
        short* pf = pd.s[tap * KC + kc].f;
        pf[0] = (short)tap; pf[1] = (short)(kc * 32);
        pf[2] = (short)tap; pf[3] = (short)(kc * 32 + 16);
      }
  };
  Desc pup1{}; build_up_prep(pup1, 2);   // CI=64
  Desc pup0{}; build_up_prep(pup0, 1);   // CI=32
  Desc cup1{}; cup1.n = 2;
  for (int kc = 0; kc < 2; ++kc) {
    short* cf = cup1.s[kc].f;
    cf[0] = 0; cf[1] = 0; cf[2] = 0; cf[3] = (short)(kc * 32);
    cf[4] = 0; cf[5] = 0; cf[6] = 0; cf[7] = (short)(kc * 32 + 16);
  }
  Desc cup0{}; cup0.n = 1;
  { short* cf = cup0.s[0].f;
    cf[0] = 0; cf[1] = 0; cf[2] = 0; cf[3] = 0;
    cf[4] = 0; cf[5] = 0; cf[6] = 0; cf[7] = 16; }

  // ----------------------------- launchers ----------------------------------
  auto prep = [&](const float* w, bf16_t* frag, int CI, int CO, int nc, const Desc& pd) {
    int total = pd.n * nc * 512;
    k_prep_w<<<(total + 255) / 256, 256, 0, stream>>>(w, frag, CI, CO, nc, pd);
  };
  auto conv = [&](const bf16_t* in, const bf16_t* wf, const float* mask,
                  const bf16_t* res, bf16_t* out,
                  int Din, int Hin, int Win, int Cin,
                  int Dc, int Hc, int Wt, int Hout, int Wout, int Cout,
                  int sh, int sw, int osh, int osw, int ooh, int oow,
                  int nc, int flags, const Desc& d) {
    dim3 grid(Wt / 4, Hc, Dc), block(128);
    switch (nc) {
      case 1:
        k_conv_wmma<1><<<grid, block, 0, stream>>>(in, wf, mask, res, out,
            Din, Hin, Win, Cin, Hout, Wout, Cout,
            sh, sw, osh, osw, ooh, oow, flags, d);
        break;
      case 2:
        k_conv_wmma<2><<<grid, block, 0, stream>>>(in, wf, mask, res, out,
            Din, Hin, Win, Cin, Hout, Wout, Cout,
            sh, sw, osh, osw, ooh, oow, flags, d);
        break;
      case 4:
        k_conv_wmma<4><<<grid, block, 0, stream>>>(in, wf, mask, res, out,
            Din, Hin, Win, Cin, Hout, Wout, Cout,
            sh, sw, osh, osw, ooh, oow, flags, d);
        break;
      default:
        k_conv_wmma<6><<<grid, block, 0, stream>>>(in, wf, mask, res, out,
            Din, Hin, Win, Cin, Hout, Wout, Cout,
            sh, sw, osh, osw, ooh, oow, flags, d);
        break;
    }
  };

  // weight prep (26 tensors)
  for (int i = 0; i < 4; ++i) prep(W[i],      fr_l0dn[i], 16, 16, 1, p3_16);
  prep(W[4],  fr_ds16, 16, 32, 2, pds16);
  for (int i = 0; i < 4; ++i) prep(W[5 + i],  fr_l1dn[i], 32, 32, 2, p3_32);
  prep(W[9],  fr_ds32, 32, 64, 4, pds32);
  prep(W[10], fr_mg,   64, 96, 6, pmg);
  for (int i = 0; i < 4; ++i) prep(W[11 + i], fr_deep[i], 96, 96, 6, p3_96);
  prep(W[15], fr_sp,   96, 64, 4, psp);
  prep(W[16], fr_up1,  64, 32, 2, pup1);
  for (int i = 0; i < 4; ++i) prep(W[17 + i], fr_l1up[i], 32, 32, 2, p3_32);
  prep(W[21], fr_up0,  32, 16, 1, pup0);
  for (int i = 0; i < 4; ++i) prep(W[22 + i], fr_l0up[i], 16, 16, 1, p3_16);

  // input convert + masks
  k_f2b<<<(int)((nL0 + 255) / 256), 256, 0, stream>>>(x, A0, nL0);
  k_pool2<<<(int)((nM1 + 255) / 256), 256, 0, stream>>>(m0, M1, 3, 256, 384, 512, 768);
  k_pool2<<<(int)((nM2 + 255) / 256), 256, 0, stream>>>(M1, M2, 3, 128, 192, 256, 384);
  k_maxd <<<(int)((nMM + 255) / 256), 256, 0, stream>>>(M2, MM, 3, 128 * 192);

  // ------------------------------ forward -----------------------------------
  // l0_down (2 res blocks, C=16, 512x768)
  for (int b = 0; b < 2; ++b) {
    conv(A0, fr_l0dn[2 * b + 0], m0, nullptr, T0, 3, 512, 768, 16,
         3, 512, 48, 512, 768, 16, 1, 1, 1, 1, 0, 0, 1, F_LEAKY, c3_16);
    conv(T0, fr_l0dn[2 * b + 1], m0, A0, A0, 3, 512, 768, 16,
         3, 512, 48, 512, 768, 16, 1, 1, 1, 1, 0, 0, 1, F_LEAKY | F_RES_BEFORE, c3_16);
  }
  // l0_ds: 16 -> 32, stride 2
  conv(A0, fr_ds16, M1, nullptr, A1, 3, 512, 768, 16,
       3, 256, 24, 256, 384, 32, 2, 2, 1, 1, 0, 0, 2, F_LEAKY, cds16);
  // l1_down
  for (int b = 0; b < 2; ++b) {
    conv(A1, fr_l1dn[2 * b + 0], M1, nullptr, T1, 3, 256, 384, 32,
         3, 256, 24, 256, 384, 32, 1, 1, 1, 1, 0, 0, 2, F_LEAKY, c3_32);
    conv(T1, fr_l1dn[2 * b + 1], M1, A1, A1, 3, 256, 384, 32,
         3, 256, 24, 256, 384, 32, 1, 1, 1, 1, 0, 0, 2, F_LEAKY | F_RES_BEFORE, c3_32);
  }
  // l1_ds: 32 -> 64, stride 2
  conv(A1, fr_ds32, M2, nullptr, B2, 3, 256, 384, 32,
       3, 128, 12, 128, 192, 64, 2, 2, 1, 1, 0, 0, 4, F_LEAKY, cds32);
  // merger: D=3 -> 1, 64 -> 96
  conv(B2, fr_mg, MM, nullptr, DP, 3, 128, 192, 64,
       1, 128, 12, 128, 192, 96, 1, 1, 1, 1, 0, 0, 6, 0, cmg);
  // deep (2 res blocks, C=96, D=1)
  for (int b = 0; b < 2; ++b) {
    conv(DP, fr_deep[2 * b + 0], MM, nullptr, TD, 1, 128, 192, 96,
         1, 128, 12, 128, 192, 96, 1, 1, 1, 1, 0, 0, 6, F_LEAKY, c3_96);
    conv(TD, fr_deep[2 * b + 1], MM, DP, DP, 1, 128, 192, 96,
         1, 128, 12, 128, 192, 96, 1, 1, 1, 1, 0, 0, 6, F_LEAKY | F_RES_BEFORE, c3_96);
  }
  // splitter: D=1 -> 3, 96 -> 64 (per-plane launches); S2 = classification head
  for (int p = 0; p < 3; ++p)
    conv(DP, fr_sp + (size_t)p * 3 * 4 * 512, M2 + (size_t)p * 128 * 192, nullptr,
         S2 + (size_t)p * 128 * 192 * 64, 1, 128, 192, 96,
         1, 128, 12, 128, 192, 64, 1, 1, 1, 1, 0, 0, 4, 0, csp);
  // l1 up: deconv 2x2 (64->32), + r1 (A1), then res blocks
  for (int py = 0; py < 2; ++py)
    for (int px = 0; px < 2; ++px) {
      int tap = py * 2 + px;
      conv(S2, fr_up1 + (size_t)tap * 2 * 2 * 512, M1, A1, A1, 3, 128, 192, 64,
           3, 128, 12, 256, 384, 32, 1, 1, 2, 2, py, px, 2,
           F_LEAKY | F_RES_AFTER, cup1);
    }
  for (int b = 0; b < 2; ++b) {
    conv(A1, fr_l1up[2 * b + 0], M1, nullptr, T1, 3, 256, 384, 32,
         3, 256, 24, 256, 384, 32, 1, 1, 1, 1, 0, 0, 2, F_LEAKY, c3_32);
    conv(T1, fr_l1up[2 * b + 1], M1, A1, A1, 3, 256, 384, 32,
         3, 256, 24, 256, 384, 32, 1, 1, 1, 1, 0, 0, 2, F_LEAKY | F_RES_BEFORE, c3_32);
  }
  // A1 is now the vertex head.
  // l0 up: deconv 2x2 (32->16), + r0 (A0), then res blocks
  for (int py = 0; py < 2; ++py)
    for (int px = 0; px < 2; ++px) {
      int tap = py * 2 + px;
      conv(A1, fr_up0 + (size_t)tap * 512, m0, A0, A0, 3, 256, 384, 32,
           3, 256, 24, 512, 768, 16, 1, 1, 2, 2, py, px, 1,
           F_LEAKY | F_RES_AFTER, cup0);
    }
  for (int b = 0; b < 2; ++b) {
    conv(A0, fr_l0up[2 * b + 0], m0, nullptr, T0, 3, 512, 768, 16,
         3, 512, 48, 512, 768, 16, 1, 1, 1, 1, 0, 0, 1, F_LEAKY, c3_16);
    conv(T0, fr_l0up[2 * b + 1], m0, A0, A0, 3, 512, 768, 16,
         3, 512, 48, 512, 768, 16, 1, 1, 1, 1, 0, 0, 1, F_LEAKY | F_RES_BEFORE, c3_16);
  }

  // ------------------------------ outputs -----------------------------------
  float* out = (float*)d_out;
  k_b2f<<<(int)((nL0 + 255) / 256), 256, 0, stream>>>(A0, out, nL0);
  k_b2f<<<(int)((nL2 + 255) / 256), 256, 0, stream>>>(S2, out + nL0, nL2);
  k_b2f<<<(int)((nL1 + 255) / 256), 256, 0, stream>>>(A1, out + nL0 + nL2, nL1);
}
